// MesaUnit_12343736009146
// MI455X (gfx1250) — matrix operand order repaired and checked by reading, not compiled, not run on hardware
//
#include <hip/hip_runtime.h>
#include <hip/hip_bf16.h>
#include <math.h>

#define B_   2
#define N_   2048
#define D_   1024
#define H_   16
#define DH_  64
#define R_   (B_ * N_)      // 4096 token rows
#define CGI_ 30

typedef __attribute__((ext_vector_type(16))) _Float16 v16h;
typedef __attribute__((ext_vector_type(8)))  _Float16 v8h;
typedef __attribute__((ext_vector_type(8)))  float    v8f;
typedef __attribute__((ext_vector_type(4)))  unsigned int v4u;
typedef __attribute__((ext_vector_type(8)))  int v8i;
typedef __attribute__((ext_vector_type(4)))  int v4i;

__device__ __forceinline__ float wred32(float v) {
#pragma unroll
  for (int m = 16; m; m >>= 1) v += __shfl_xor(v, m, 32);
  return v;
}

// ---------------- f32 -> f16 convert ----------------
__global__ __launch_bounds__(256) void cvt_f16_kernel(const float* __restrict__ src,
                                                      _Float16* __restrict__ dst, int n) {
  int i = blockIdx.x * 256 + threadIdx.x;
  if (i < n) dst[i] = (_Float16)src[i];
}

// ---------------- pack [Wf | Wbet | 0] into a 1024x64 f16 tile ----------------
__global__ __launch_bounds__(256) void pack_wfb_kernel(const float* __restrict__ Wf,
                                                       const float* __restrict__ Wbet,
                                                       _Float16* __restrict__ Wfbh) {
  int i = blockIdx.x * 256 + threadIdx.x;
  if (i >= D_ * 64) return;
  int row = i >> 6, col = i & 63;
  float v = 0.f;
  if (col < 16)      v = Wf[row * 16 + col];
  else if (col < 32) v = Wbet[row * 16 + (col - 16)];
  Wfbh[i] = (_Float16)v;
}

// ---------------- TDM: DMA a TMxTK f16 tile (row stride K halfs) into LDS ----
// D# group0: count=1 | lds_addr | global_addr(57b) | type=2
// D# group1: data_size=2B | tensor_dim0/1 | tile_dim0/1 | tensor_dim0_stride=K
__device__ __forceinline__ void tdm_load_tile_f16(const _Float16* gsrc, int rowStrideElems,
                                                  int tileW, int tileH, _Float16* ldsDst) {
  unsigned long long ga = (unsigned long long)(const void*)gsrc;
  unsigned int la = (unsigned int)(unsigned long long)(void*)ldsDst;  // LDS byte offset
  v4u g0;
  g0.x = 1u;                                            // count=1, user descriptor
  g0.y = la;                                            // lds_addr
  g0.z = (unsigned int)(ga & 0xffffffffull);            // global_addr[31:0]
  g0.w = (unsigned int)((ga >> 32) & 0x1ffffffull)      // global_addr[56:32]
         | (2u << 30);                                  // type=2 (image)
  unsigned int tw = (unsigned int)tileW, thh = (unsigned int)tileH;
  unsigned int st = (unsigned int)rowStrideElems;
  v8i g1;
  g1[0] = (int)(1u << 16);                              // data_size=1 -> 2 bytes
  g1[1] = (int)((tw & 0xffffu) << 16);                  // tensor_dim0[15:0]
  g1[2] = (int)(((tw >> 16) & 0xffffu) | ((thh & 0xffffu) << 16)); // dim0 hi | dim1 lo
  g1[3] = (int)(((thh >> 16) & 0xffffu) | (tw << 16));  // dim1 hi | tile_dim0
  g1[4] = (int)(thh & 0xffffu);                         // tile_dim1 | tile_dim2=0
  g1[5] = (int)st;                                      // tensor_dim0_stride[31:0]
  g1[6] = 0;                                            // stride hi | dim1_stride lo
  g1[7] = 0;
  v4i gz4 = {0, 0, 0, 0};                               // 2-D tile: groups 2/3 unused
  v8i gz8 = {0, 0, 0, 0, 0, 0, 0, 0};
  __builtin_amdgcn_tensor_load_to_lds(g0, g1, gz4, gz4, gz8, 0);
}

// ---------------- WMMA GEMM: C[M,N] = A[M,K] * B[K,N], f16 in / f32 out ----
// tile: 128(M) x 64(N), K-step 32; 8 waves; A staged by TDM, B fragment-swizzled;
// double-buffered K pipeline, one barrier per K-step.
#define TM 128
#define TN 64
#define TK 32
__global__ __launch_bounds__(256) void gemm_wmma_kernel(const _Float16* __restrict__ A,
                                                        const _Float16* __restrict__ Bw,
                                                        float* __restrict__ C,
                                                        int M, int N, int K) {
  __shared__ __align__(32) _Float16 As[2][TM * TK];     // 2 x 8 KB
  __shared__ __align__(32) _Float16 Bsw[2][4][32][16];  // 2 x 4 KB, fragment-swizzled

  const int tid   = threadIdx.x;
  const int wave  = tid >> 5;
  const int lane  = tid & 31;
  const int half  = lane >> 4;
  const int mfrag = lane & 15;
  const int bm = blockIdx.x * TM;
  const int bn = blockIdx.y * TN;

  v8f acc[4];
#pragma unroll
  for (int c = 0; c < 4; ++c)
#pragma unroll
    for (int r = 0; r < 8; ++r) acc[c][r] = 0.f;

  const int brow = tid >> 3, bseg = tid & 7;   // each thread: 8 cols of one B row

  // ---- prologue: stage tile 0 ----
  if (wave == 0)
    tdm_load_tile_f16(A + (size_t)bm * K, K, TK, TM, &As[0][0]);
  {
    v8h bv = *(const v8h*)(Bw + (size_t)brow * N + bn + bseg * 8);
#pragma unroll
    for (int j = 0; j < 8; ++j) {
      int col = bseg * 8 + j;
      Bsw[0][col >> 4][(brow >> 4) * 16 + (col & 15)][brow & 15] = bv[j];
    }
  }
  if (wave == 0) __builtin_amdgcn_s_wait_tensorcnt(0);
  __syncthreads();

  const int nTiles = K / TK;
  for (int it = 0; it < nTiles; ++it) {
    const int cur = it & 1;
    const int nxt = cur ^ 1;
    const bool hasNext = (it + 1) < nTiles;

    v8h bnv;
    if (hasNext) {
      const int k0n = (it + 1) * TK;
      if (wave == 0)
        tdm_load_tile_f16(A + (size_t)bm * K + k0n, K, TK, TM, &As[nxt][0]);
      bnv = *(const v8h*)(Bw + (size_t)(k0n + brow) * N + bn + bseg * 8);
    }

    // ---- A fragment: 16x32, lanes 0-15 K[0..15], lanes 16-31 K[16..31]
    const v8h* arp = (const v8h*)(&As[cur][(wave * 16 + mfrag) * TK]);
    v8h alo = arp[2 * half];
    v8h ahi = arp[2 * half + 1];
    v16h af;
#pragma unroll
    for (int j = 0; j < 8; ++j) { af[j] = alo[j]; af[8 + j] = ahi[j]; }

#pragma unroll
    for (int c = 0; c < 4; ++c) {
      const v8h* brp = (const v8h*)(&Bsw[cur][c][lane][0]);
      v8h blo = brp[0], bhi = brp[1];
      v16h bf;
#pragma unroll
      for (int j = 0; j < 8; ++j) { bf[j] = blo[j]; bf[8 + j] = bhi[j]; }
      acc[c] = __builtin_amdgcn_wmma_f32_16x16x32_f16(false, af, false, bf,
                                                      (short)0, acc[c], false, false);
    }

    if (hasNext) {
#pragma unroll
      for (int j = 0; j < 8; ++j) {
        int col = bseg * 8 + j;
        Bsw[nxt][col >> 4][(brow >> 4) * 16 + (col & 15)][brow & 15] = bnv[j];
      }
      if (wave == 0) __builtin_amdgcn_s_wait_tensorcnt(0);
      __syncthreads();
    }
  }

  // ---- store D: VGPR r -> row (half*8 + r), col = lane&15
#pragma unroll
  for (int c = 0; c < 4; ++c)
#pragma unroll
    for (int r = 0; r < 8; ++r) {
      int row = bm + wave * 16 + half * 8 + r;
      int col = bn + c * 16 + mfrag;
      C[(size_t)row * N + col] = acc[c][r];
    }
}

// ---------------- per-head L2 norm of q and k (in place) ----------------
__global__ __launch_bounds__(32) void l2norm_kernel(float* __restrict__ q,
                                                    float* __restrict__ k) {
  size_t base = (size_t)blockIdx.x * 64;
  int lane = threadIdx.x;
  float a0 = q[base + lane], a1 = q[base + lane + 32];
  float rq = rsqrtf(wred32(a0 * a0 + a1 * a1) + 1e-6f);
  q[base + lane] = a0 * rq; q[base + lane + 32] = a1 * rq;
  float b0 = k[base + lane], b1 = k[base + lane + 32];
  float rk = rsqrtf(wred32(b0 * b0 + b1 * b1) + 1e-6f);
  k[base + lane] = b0 * rk; k[base + lane + 32] = b1 * rk;
}

// ---------------- log_f / beta from the packed f|beta GEMM output ----------
__global__ __launch_bounds__(256) void fbeta_kernel(const float* __restrict__ fbf,
                                                    const float* __restrict__ delta,
                                                    float* __restrict__ logf,
                                                    float* __restrict__ beta) {
  int i = blockIdx.x * 256 + threadIdx.x;
  if (i >= R_ * H_) return;
  int row = i / H_, h = i % H_;
  float z = fbf[row * 64 + h] + delta[h];
  float m = fminf(z, 0.f);
  logf[i] = m - log1pf(__expf(-fabsf(z)));      // stable log-sigmoid
  float zb = fbf[row * 64 + 16 + h];
  beta[i] = 1.f / (1.f + __expf(-zb));
}

// ---------------- lamb = softplus(p) + 0.25 ----------------
__global__ __launch_bounds__(256) void lamb_kernel(const float* __restrict__ p,
                                                   float* __restrict__ lamb) {
  int i = blockIdx.x * 256 + threadIdx.x;
  if (i >= D_) return;
  float z = p[i];
  lamb[i] = fmaxf(z, 0.f) + log1pf(__expf(-fabsf(z))) + 0.25f;
}

// ---------------- recurrent scan: one wave32 per (b,h) chain ----------------
// lane L owns vector elements / matrix rows {L, L+32}; all CG reductions are
// in-wave shuffles; H_kk/H_kv live in LDS (32 KB; 33 KB total with pv/kvs/vvs/xs, of the 320 KB WGP pool).
__global__ __launch_bounds__(32) void scan_kernel(const float* __restrict__ qn,
                                                  const float* __restrict__ kn,
                                                  const float* __restrict__ vn,
                                                  const float* __restrict__ logf,
                                                  const float* __restrict__ beta,
                                                  const float* __restrict__ lamb,
                                                  float* __restrict__ o) {
  __shared__ __align__(16) float Hkk[64 * 64];
  __shared__ __align__(16) float Hkv[64 * 64];
  __shared__ __align__(16) float pv[64];
  __shared__ __align__(16) float kvs[64];
  __shared__ __align__(16) float vvs[64];
  __shared__ __align__(16) float xs[64];

  const int lane = threadIdx.x;
  const int b = blockIdx.x / H_, h = blockIdx.x % H_;
  const int r0 = lane, r1 = lane + 32;

  for (int i = lane; i < 64 * 64; i += 32) { Hkk[i] = 0.f; Hkv[i] = 0.f; }
  const float lam0 = lamb[h * 64 + r0];
  const float lam1 = lamb[h * 64 + r1];
  __syncthreads();

  for (int t = 0; t < N_; ++t) {
    const size_t base = ((size_t)(b * N_ + t) * H_ + h) * 64;
    const size_t sb   = (size_t)(b * N_ + t) * H_ + h;
    float q0 = qn[base + r0], q1 = qn[base + r1];
    float k0 = kn[base + r0], k1 = kn[base + r1];
    float v0 = vn[base + r0], v1 = vn[base + r1];
    float fexp = __expf(logf[sb]);
    float bt = beta[sb];

    // latency-bound loop: pull next token's operands toward the caches now
    if (t + 1 < N_) {
      const size_t nb = base + (size_t)H_ * 64;
      __builtin_prefetch(qn + nb + r0, 0, 1);
      __builtin_prefetch(kn + nb + r0, 0, 1);
      __builtin_prefetch(vn + nb + r0, 0, 1);
    }

    kvs[r0] = k0; kvs[r1] = k1;
    vvs[r0] = v0; vvs[r1] = v1;
    pv[r0] = q0;  pv[r1] = q1;
    __syncthreads();

    // H_kk = f*H_kk + (beta k)k^T ;  H_kv = f*H_kv + (beta k)v^T
    const float bk0 = bt * k0, bk1 = bt * k1;
#pragma unroll 4
    for (int j = 0; j < 64; j += 4) {
      float4 kj = *(const float4*)&kvs[j];
      float4 vj = *(const float4*)&vvs[j];
      float4 a0 = *(float4*)&Hkk[r0 * 64 + j];
      a0.x = fexp * a0.x + bk0 * kj.x; a0.y = fexp * a0.y + bk0 * kj.y;
      a0.z = fexp * a0.z + bk0 * kj.z; a0.w = fexp * a0.w + bk0 * kj.w;
      *(float4*)&Hkk[r0 * 64 + j] = a0;
      float4 a1 = *(float4*)&Hkk[r1 * 64 + j];
      a1.x = fexp * a1.x + bk1 * kj.x; a1.y = fexp * a1.y + bk1 * kj.y;
      a1.z = fexp * a1.z + bk1 * kj.z; a1.w = fexp * a1.w + bk1 * kj.w;
      *(float4*)&Hkk[r1 * 64 + j] = a1;
      float4 c0 = *(float4*)&Hkv[r0 * 64 + j];
      c0.x = fexp * c0.x + bk0 * vj.x; c0.y = fexp * c0.y + bk0 * vj.y;
      c0.z = fexp * c0.z + bk0 * vj.z; c0.w = fexp * c0.w + bk0 * vj.w;
      *(float4*)&Hkv[r0 * 64 + j] = c0;
      float4 c1 = *(float4*)&Hkv[r1 * 64 + j];
      c1.x = fexp * c1.x + bk1 * vj.x; c1.y = fexp * c1.y + bk1 * vj.y;
      c1.z = fexp * c1.z + bk1 * vj.z; c1.w = fexp * c1.w + bk1 * vj.w;
      *(float4*)&Hkv[r1 * 64 + j] = c1;
    }
    __syncthreads();

    // CG on (H_kk + diag(lamb)) x = q
    float x0 = 0.f, x1 = 0.f;
    float rr0 = q0, rr1 = q1;
    float p0 = q0, p1 = q1;
    float rs = wred32(q0 * q0 + q1 * q1);
    for (int it = 0; it < CGI_; ++it) {
      float ap0 = lam0 * p0, ap1 = lam1 * p1;
#pragma unroll 4
      for (int j = 0; j < 64; j += 4) {
        float4 pj = *(const float4*)&pv[j];
        float4 a0 = *(const float4*)&Hkk[r0 * 64 + j];
        float4 a1 = *(const float4*)&Hkk[r1 * 64 + j];
        ap0 += a0.x * pj.x + a0.y * pj.y + a0.z * pj.z + a0.w * pj.w;
        ap1 += a1.x * pj.x + a1.y * pj.y + a1.z * pj.z + a1.w * pj.w;
      }
      float pAp = wred32(p0 * ap0 + p1 * ap1);
      float alpha = rs / (pAp + 1e-12f);
      x0 += alpha * p0;  x1 += alpha * p1;
      rr0 -= alpha * ap0; rr1 -= alpha * ap1;
      float rsn = wred32(rr0 * rr0 + rr1 * rr1);
      float bcg = rsn / (rs + 1e-12f);
      p0 = rr0 + bcg * p0; p1 = rr1 + bcg * p1;
      rs = rsn;
      __syncthreads();
      pv[r0] = p0; pv[r1] = p1;
      __syncthreads();
    }

    xs[r0] = x0; xs[r1] = x1;
    __syncthreads();
    // o_t = x^T H_kv (column reduction, bank-conflict-free: bank = lane)
    float o0 = 0.f, o1 = 0.f;
#pragma unroll 8
    for (int i = 0; i < 64; ++i) {
      float xi = xs[i];
      o0 += xi * Hkv[i * 64 + r0];
      o1 += xi * Hkv[i * 64 + r1];
    }
    o[base + r0] = o0; o[base + r1] = o1;
    __syncthreads();
  }
}

// ---------------- gate * o, grouped RMSNorm, weight, -> f16 ----------------
__global__ __launch_bounds__(32) void gate_norm_kernel(const float* __restrict__ obuf,
                                                       const float* __restrict__ gbuf,
                                                       const float* __restrict__ nw,
                                                       _Float16* __restrict__ ogh) {
  int g = blockIdx.x;          // over R_*H_ groups
  int lane = threadIdx.x;
  int row = g / H_, h = g % H_;
  size_t ob = (size_t)g * 64;
  size_t gb = (size_t)row * D_ + h * 64;
  float o0 = obuf[ob + lane], o1 = obuf[ob + lane + 32];
  float g0 = gbuf[gb + lane], g1 = gbuf[gb + lane + 32];
  o0 *= 1.f / (1.f + __expf(-g0));
  o1 *= 1.f / (1.f + __expf(-g1));
  float ms = wred32(o0 * o0 + o1 * o1) * (1.f / 64.f);
  float r = rsqrtf(ms + 1e-5f);
  ogh[gb + lane]      = (_Float16)(o0 * r * nw[h * 64 + lane]);
  ogh[gb + lane + 32] = (_Float16)(o1 * r * nw[h * 64 + lane + 32]);
}

// ------------------------------------------------------------------
extern "C" void kernel_launch(void* const* d_in, const int* in_sizes, int n_in,
                              void* d_out, int out_size, void* d_ws, size_t ws_size,
                              hipStream_t stream) {
  const float* x     = (const float*)d_in[0];
  const float* Wq    = (const float*)d_in[1];
  const float* Wk    = (const float*)d_in[2];
  const float* Wv    = (const float*)d_in[3];
  const float* Wf    = (const float*)d_in[4];
  const float* Wbet  = (const float*)d_in[5];
  const float* Wo    = (const float*)d_in[6];
  const float* delta = (const float*)d_in[7];
  const float* lambp = (const float*)d_in[8];
  const float* nw    = (const float*)d_in[9];
  const float* gw1   = (const float*)d_in[10];
  const float* gw2   = (const float*)d_in[11];
  float* out = (float*)d_out;

  char* ws = (char*)d_ws;
  size_t off = 0;
  auto take = [&](size_t bytes) -> char* {
    char* p = ws + off;
    off = (off + bytes + 255) & ~(size_t)255;
    return p;
  };

  _Float16* xh   = (_Float16*)take((size_t)R_ * D_ * 2);
  _Float16* Wqh  = (_Float16*)take((size_t)D_ * D_ * 2);
  _Float16* Wkh  = (_Float16*)take((size_t)D_ * D_ * 2);
  _Float16* Wvh  = (_Float16*)take((size_t)D_ * D_ * 2);
  _Float16* Woh  = (_Float16*)take((size_t)D_ * D_ * 2);
  _Float16* Wfbh = (_Float16*)take((size_t)D_ * 64 * 2);
  _Float16* gw1h = (_Float16*)take((size_t)D_ * 64 * 2);
  _Float16* gw2h = (_Float16*)take((size_t)64 * D_ * 2);
  float* qf   = (float*)take((size_t)R_ * D_ * 4);
  float* kf   = (float*)take((size_t)R_ * D_ * 4);
  float* vf   = (float*)take((size_t)R_ * D_ * 4);
  float* fbf  = (float*)take((size_t)R_ * 64 * 4);
  float* xg1f = (float*)take((size_t)R_ * 64 * 4);
  _Float16* xg1h = (_Float16*)take((size_t)R_ * 64 * 2);
  float* logf = (float*)take((size_t)R_ * H_ * 4);
  float* beta = (float*)take((size_t)R_ * H_ * 4);
  float* lamb = (float*)take((size_t)D_ * 4);
  float* gbuf = (float*)take((size_t)R_ * D_ * 4);
  float* obuf = (float*)take((size_t)R_ * D_ * 4);
  _Float16* ogh = (_Float16*)take((size_t)R_ * D_ * 2);

  // 1) f32 -> f16 conversions
  cvt_f16_kernel<<<(R_ * D_) / 256, 256, 0, stream>>>(x, xh, R_ * D_);
  cvt_f16_kernel<<<(D_ * D_) / 256, 256, 0, stream>>>(Wq, Wqh, D_ * D_);
  cvt_f16_kernel<<<(D_ * D_) / 256, 256, 0, stream>>>(Wk, Wkh, D_ * D_);
  cvt_f16_kernel<<<(D_ * D_) / 256, 256, 0, stream>>>(Wv, Wvh, D_ * D_);
  cvt_f16_kernel<<<(D_ * D_) / 256, 256, 0, stream>>>(Wo, Woh, D_ * D_);
  cvt_f16_kernel<<<(D_ * 64) / 256, 256, 0, stream>>>(gw1, gw1h, D_ * 64);
  cvt_f16_kernel<<<(64 * D_) / 256, 256, 0, stream>>>(gw2, gw2h, 64 * D_);
  pack_wfb_kernel<<<(D_ * 64) / 256, 256, 0, stream>>>(Wf, Wbet, Wfbh);

  // 2) WMMA projections (A tiles staged by TDM, double-buffered)
  dim3 gQ(R_ / TM, D_ / TN);
  gemm_wmma_kernel<<<gQ, 256, 0, stream>>>(xh, Wqh, qf, R_, D_, D_);
  gemm_wmma_kernel<<<gQ, 256, 0, stream>>>(xh, Wkh, kf, R_, D_, D_);
  gemm_wmma_kernel<<<gQ, 256, 0, stream>>>(xh, Wvh, vf, R_, D_, D_);
  dim3 gS(R_ / TM, 1);
  gemm_wmma_kernel<<<gS, 256, 0, stream>>>(xh, Wfbh, fbf, R_, 64, D_);
  gemm_wmma_kernel<<<gS, 256, 0, stream>>>(xh, gw1h, xg1f, R_, 64, D_);

  // 3) pointwise prep
  l2norm_kernel<<<R_ * H_, 32, 0, stream>>>(qf, kf);
  fbeta_kernel<<<(R_ * H_) / 256, 256, 0, stream>>>(fbf, delta, logf, beta);
  lamb_kernel<<<(D_ + 255) / 256, 256, 0, stream>>>(lambp, lamb);

  // 4) gate second matmul (K=64) via WMMA
  cvt_f16_kernel<<<(R_ * 64) / 256, 256, 0, stream>>>(xg1f, xg1h, R_ * 64);
  gemm_wmma_kernel<<<gQ, 256, 0, stream>>>(xg1h, gw2h, gbuf, R_, D_, 64);

  // 5) sequential scan: 32 chains, one wave32 each
  scan_kernel<<<B_ * H_, 32, 0, stream>>>(qf, kf, vf, logf, beta, lamb, obuf);

  // 6) gate + grouped RMSNorm -> f16
  gate_norm_kernel<<<R_ * H_, 32, 0, stream>>>(obuf, gbuf, nw, ogh);

  // 7) output projection via WMMA
  gemm_wmma_kernel<<<gQ, 256, 0, stream>>>(ogh, Woh, out, R_, D_, D_);
}
